// GATReceptionPredictor_41558103556531
// MI455X (gfx1250) — matrix operand order repaired and checked by reading, not compiled, not run on hardware
//
#include <hip/hip_runtime.h>
#include <hip/hip_bf16.h>
#include <math.h>

typedef __attribute__((ext_vector_type(16))) _Float16 v16h;
typedef __attribute__((ext_vector_type(8)))  _Float16 v8h;
typedef __attribute__((ext_vector_type(2)))  _Float16 v2h;
typedef __attribute__((ext_vector_type(8)))  float    v8f;

#define N_NODES 50000
#define N_EDGES 800000
#define EP      (N_EDGES + N_NODES)   // edges incl. self loops
#define NFEAT   64
#define EFEAT   12
#define HID     128
#define EH      32
#define NHEAD   8
#define CPH     16
#define NEG_SLOPE 0.2f

// ---- monotone uint encoding of f32 for atomicMax-based segment max ----
__device__ __forceinline__ unsigned enc_f32(float f) {
  unsigned u = __float_as_uint(f);
  return (u & 0x80000000u) ? ~u : (u | 0x80000000u);
}
__device__ __forceinline__ float dec_f32(unsigned u) {
  unsigned b = (u & 0x80000000u) ? (u & 0x7FFFFFFFu) : ~u;
  return __uint_as_float(b);
}

// ---------------- generic fill ----------------
__global__ void k_fill(float* __restrict__ p, long n, float v) {
  long i = (long)blockIdx.x * blockDim.x + threadIdx.x;
  if (i < n) p[i] = v;
}

// ---------------- edge encoder: ef2[e,j] = ee_b[j] + edge_attr[e,:12] . ee_W[j,:12] ----------------
__global__ void k_edge_enc(const float* __restrict__ ea, const float* __restrict__ W,
                           const float* __restrict__ b, float* __restrict__ ef2) {
  long t = (long)blockIdx.x * blockDim.x + threadIdx.x;
  if (t >= (long)N_EDGES * EH) return;
  int e = (int)(t >> 5), j = (int)(t & 31);
  const float* ar = ea + (size_t)e * EFEAT;
  const float* wr = W + (size_t)j * EFEAT;
  float acc = b[j];
#pragma unroll
  for (int k = 0; k < EFEAT; ++k) acc = fmaf(ar[k], wr[k], acc);
  ef2[(size_t)e * EH + j] = acc;
}

// ---------------- in-degree ----------------
__global__ void k_deg(const int* __restrict__ ei, float* __restrict__ deg) {
  long e = (long)blockIdx.x * blockDim.x + threadIdx.x;
  if (e >= N_EDGES) return;
  atomicAdd(&deg[ei[N_EDGES + e]], 1.0f);
}

// ---------------- accumulate incoming edge feats into self-loop rows ----------------
__global__ void k_loop_acc(const int* __restrict__ ei, float* __restrict__ ef2) {
  long t = (long)blockIdx.x * blockDim.x + threadIdx.x;
  if (t >= (long)N_EDGES * EH) return;
  int e = (int)(t >> 5), j = (int)(t & 31);
  int d = ei[N_EDGES + e];
  atomicAdd(&ef2[((size_t)N_EDGES + d) * EH + j], ef2[(size_t)e * EH + j]);
}

__global__ void k_loop_div(const float* __restrict__ deg, float* __restrict__ ef2) {
  long t = (long)blockIdx.x * blockDim.x + threadIdx.x;
  if (t >= (long)N_NODES * EH) return;
  int n = (int)(t >> 5), j = (int)(t & 31);
  ef2[((size_t)N_EDGES + n) * EH + j] /= fmaxf(deg[n], 1.0f);
}

// ---------------- WMMA GEMM: Out[M,128] = A[M,K] * W[128,K]^T (+bias)(+relu) ----------------
// A can be a concat of two sources (A0 for k<ksplit, A1 for k>=ksplit), ksplit % 32 == 0,
// so the source select is wave-uniform per k-step (no divergence -> EXEC all-ones at WMMA).
// LDS double-buffered tiles: A 16x32 f16, B (weights) 128x32 f16 shared by all 8 waves.
// Fragment reads are contiguous ds_load_b128s per the ISA 7.12.2 wave32 layouts.
__global__ void k_gemm_wmma(const float* __restrict__ A0, int lda0,
                            const float* __restrict__ A1, int ksplit,
                            const float* __restrict__ W, int K,
                            const float* __restrict__ bias,
                            float* __restrict__ Out, int doRelu) {
  __shared__ alignas(16) _Float16 Ah[2][16 * 32];    // [buf][m*32 + k]
  __shared__ alignas(16) _Float16 Bh[2][128 * 32];   // [buf][n*32 + k]

  const int tid     = threadIdx.x;
  const int lane    = tid & 31;
  const int wave    = tid >> 5;
  const int rowTile = blockIdx.x << 4;

  // cooperative-load coordinates (one f32x2 -> f16x2 pair per thread per tile row-chunk)
  const int am = tid >> 4;             // 0..15  (A row in tile)
  const int ak = (tid & 15) << 1;      // 0..30  (A col pair)
  const int bn = tid >> 1;             // 0..127 (B row = output col)
  const int bk = (tid & 1) << 4;       // 0 | 16 (B col chunk base, 8 pairs each)

  // fragment coordinates (ISA 7.12.2)
  const int mrow = rowTile + (lane & 15);
  const int kbA  = (lane < 16) ? 0 : 8;     // A half-wave K offset
  const int nW   = (wave << 4) + (lane & 15);
  const int kbB  = (lane < 16) ? 0 : 8;     // B half-wave K offset

  auto load_tile = [&](int k0, int buf) {
    const bool lo = (k0 < ksplit);
    const float* Arow = (lo ? A0 + (size_t)(rowTile + am) * lda0 + k0
                            : A1 + (size_t)(rowTile + am) * HID + (k0 - ksplit));
    float2 av = *(const float2*)(Arow + ak);
    *(v2h*)(&Ah[buf][am * 32 + ak]) = (v2h){(_Float16)av.x, (_Float16)av.y};
    const float* Wrow = W + (size_t)bn * K + k0 + bk;
#pragma unroll
    for (int p = 0; p < 8; ++p) {
      float2 wv = *(const float2*)(Wrow + 2 * p);
      *(v2h*)(&Bh[buf][bn * 32 + bk + 2 * p]) = (v2h){(_Float16)wv.x, (_Float16)wv.y};
    }
  };

  load_tile(0, 0);
  v8f acc = {};
  for (int k0 = 0; k0 < K; k0 += 32) {
    const int buf = (k0 >> 5) & 1;
    __syncthreads();                       // tile `buf` ready; prev reads of buf^1 done
    if (k0 + 32 < K) load_tile(k0 + 32, buf ^ 1);
    if (k0 + 64 < K) {                     // prefetch 2 tiles ahead -> global_prefetch_b8
      const bool lo2 = (k0 + 64 < ksplit);
      const float* pf = (lo2 ? A0 + (size_t)(rowTile + am) * lda0 + (k0 + 64)
                             : A1 + (size_t)(rowTile + am) * HID + (k0 + 64 - ksplit));
      __builtin_prefetch(pf + ak, 0, 1);
    }
    // A fragment: two contiguous 8-half chunks -> 2x ds_load_b128
    v8h alo = *(const v8h*)(&Ah[buf][(lane & 15) * 32 + kbA]);
    v8h ahi = *(const v8h*)(&Ah[buf][(lane & 15) * 32 + 16 + kbA]);
    v16h a = __builtin_shufflevector(alo, ahi, 0, 1, 2, 3, 4, 5, 6, 7,
                                     8, 9, 10, 11, 12, 13, 14, 15);
    // B fragment: two contiguous 8-half chunks of W row nW -> 2x ds_load_b128
    v8h blo = *(const v8h*)(&Bh[buf][(nW - 0) * 32 + kbB]);
    v8h bhi = *(const v8h*)(&Bh[buf][(nW - 0) * 32 + 16 + kbB]);
    v16h b = __builtin_shufflevector(blo, bhi, 0, 1, 2, 3, 4, 5, 6, 7,
                                     8, 9, 10, 11, 12, 13, 14, 15);
    acc = __builtin_amdgcn_wmma_f32_16x16x32_f16(false, a, false, b,
                                                 (short)0, acc, false, false);
  }
  const int mOff = (lane < 16) ? 0 : 8;    // C/D: VGPR r -> M=r (lo) / M=r+8 (hi)
  const float bv = bias ? bias[nW] : 0.0f;
#pragma unroll
  for (int r = 0; r < 8; ++r) {
    float v = acc[r] + bv;
    if (doRelu) v = fmaxf(v, 0.0f);
    Out[(size_t)(rowTile + mOff + r) * HID + nW] = v;
  }
}

// ---------------- per-(edge,head) GATv2 attention score + segment max ----------------
__global__ void k_scores(const int* __restrict__ ei, const float* __restrict__ ef2,
                         const float* __restrict__ xl, const float* __restrict__ xr,
                         const float* __restrict__ We,   // [HID, EH]
                         const float* __restrict__ att,  // [H, C]
                         float* __restrict__ alphaBuf,   // [EP*8] staging in d_out
                         unsigned* __restrict__ emax) {  // [N*8]
  long t = (long)blockIdx.x * blockDim.x + threadIdx.x;
  if (t >= (long)EP * NHEAD) return;
  int e = (int)(t >> 3), h = (int)(t & 7);
  int s, d;
  if (e < N_EDGES) { s = ei[e]; d = ei[N_EDGES + e]; } else { s = e - N_EDGES; d = s; }
  float efv[EH];
#pragma unroll
  for (int j = 0; j < EH; ++j) efv[j] = ef2[(size_t)e * EH + j];
  const float* xls = xl + (size_t)s * HID + h * CPH;
  const float* xrd = xr + (size_t)d * HID + h * CPH;
  const float* Wh  = We + (size_t)(h * CPH) * EH;
  const float* ah  = att + h * CPH;
  float sc = 0.0f;
#pragma unroll 4
  for (int c = 0; c < CPH; ++c) {
    float m = xls[c] + xrd[c];
    const float* wr = Wh + c * EH;
#pragma unroll
    for (int j = 0; j < EH; ++j) m = fmaf(wr[j], efv[j], m);  // on-the-fly edge proj
    m = (m > 0.0f) ? m : NEG_SLOPE * m;                       // LeakyReLU
    sc = fmaf(m, ah[c], sc);
  }
  alphaBuf[t] = sc;
  atomicMax(&emax[(size_t)d * NHEAD + h], enc_f32(sc));
}

// ---------------- exp(e - max) + segment sum ----------------
__global__ void k_expsum(const int* __restrict__ ei, float* __restrict__ alphaBuf,
                         const unsigned* __restrict__ emax, float* __restrict__ esum) {
  long t = (long)blockIdx.x * blockDim.x + threadIdx.x;
  if (t >= (long)EP * NHEAD) return;
  int e = (int)(t >> 3), h = (int)(t & 7);
  int d = (e < N_EDGES) ? ei[N_EDGES + e] : (e - N_EDGES);
  float ev = alphaBuf[t];
  float ex = __expf(ev - dec_f32(emax[(size_t)d * NHEAD + h]));
  alphaBuf[t] = ex;
  atomicAdd(&esum[(size_t)d * NHEAD + h], ex);
}

// ---------------- normalize alpha + alpha-weighted aggregation ----------------
__global__ void k_aggregate(const int* __restrict__ ei, float* __restrict__ alphaBuf,
                            const float* __restrict__ esum, const float* __restrict__ xl,
                            float* __restrict__ hout) {
  long t = (long)blockIdx.x * blockDim.x + threadIdx.x;
  if (t >= (long)EP * NHEAD) return;
  int e = (int)(t >> 3), h = (int)(t & 7);
  int s, d;
  if (e < N_EDGES) { s = ei[e]; d = ei[N_EDGES + e]; } else { s = e - N_EDGES; d = s; }
  float al = alphaBuf[t] / (esum[(size_t)d * NHEAD + h] + 1e-16f);
  alphaBuf[t] = al;   // final attention output (a1/a2)
  const float* xls = xl + (size_t)s * HID + h * CPH;
  float* op = hout + (size_t)d * HID + h * CPH;
#pragma unroll
  for (int c = 0; c < CPH; ++c) atomicAdd(&op[c], al * xls[c]);
}

// ---------------- out-of-layer bias + ReLU ----------------
__global__ void k_bias_relu(float* __restrict__ hbuf, const float* __restrict__ bias) {
  long t = (long)blockIdx.x * blockDim.x + threadIdx.x;
  if (t >= (long)N_NODES * HID) return;
  hbuf[t] = fmaxf(hbuf[t] + bias[t & (HID - 1)], 0.0f);
}

// ---------------- final MLP row: sigmoid(mhid . m2_W + b) ----------------
__global__ void k_mlp_out(const float* __restrict__ mhid, const float* __restrict__ w2,
                          const float* __restrict__ b2, float* __restrict__ out) {
  long n = (long)blockIdx.x * blockDim.x + threadIdx.x;
  if (n >= N_NODES) return;
  const float* hr = mhid + (size_t)n * HID;
  float z = b2[0];
#pragma unroll 8
  for (int k = 0; k < HID; ++k) z = fmaf(hr[k], w2[k], z);
  out[n] = 1.0f / (1.0f + __expf(-z));
}

static inline unsigned nblk(long n) { return (unsigned)((n + 255) / 256); }

extern "C" void kernel_launch(void* const* d_in, const int* in_sizes, int n_in,
                              void* d_out, int out_size, void* d_ws, size_t ws_size,
                              hipStream_t stream) {
  const float* x    = (const float*)d_in[0];
  const int*   ei   = (const int*)d_in[1];      // [2,E]
  const float* eatt = (const float*)d_in[2];
  const float* ne_W = (const float*)d_in[4];
  const float* ne_b = (const float*)d_in[5];
  const float* ee_W = (const float*)d_in[6];
  const float* ee_b = (const float*)d_in[7];
  const float* m1_W = (const float*)d_in[8];
  const float* m1_b = (const float*)d_in[9];
  const float* m2_W = (const float*)d_in[10];
  const float* m2_b = (const float*)d_in[11];
  // g1: 12..18  g2: 19..25  (Wl, bl, Wr, br, We, att, bias)
  const float* gW[2][7];
  for (int l = 0; l < 2; ++l)
    for (int j = 0; j < 7; ++j) gW[l][j] = (const float*)d_in[12 + 7 * l + j];

  float* out = (float*)d_out;
  float* alpha[2];
  alpha[0] = out + N_NODES;
  alpha[1] = alpha[0] + (size_t)EP * NHEAD;

  // ---- workspace carve-out (all f32) ----
  float* ws = (float*)d_ws;
  size_t off = 0;
  float* h0   = ws + off; off += (size_t)N_NODES * HID;
  float* ef2  = ws + off; off += (size_t)EP * EH;
  float* xl   = ws + off; off += (size_t)N_NODES * HID;
  float* xr   = ws + off; off += (size_t)N_NODES * HID;
  float* h1   = ws + off; off += (size_t)N_NODES * HID;
  float* h2   = ws + off; off += (size_t)N_NODES * HID;
  float* mhid = ws + off; off += (size_t)N_NODES * HID;
  float* deg  = ws + off; off += (size_t)N_NODES;
  float* emaxf= ws + off; off += (size_t)N_NODES * NHEAD;
  float* esum = ws + off; off += (size_t)N_NODES * NHEAD;
  unsigned* emax = (unsigned*)emaxf;

  const dim3 B(256);

  // --- edge encoder + self-loop mean edge features ---
  k_fill<<<nblk((long)N_NODES * EH), B, 0, stream>>>(ef2 + (size_t)N_EDGES * EH,
                                                     (long)N_NODES * EH, 0.0f);
  k_fill<<<nblk(N_NODES), B, 0, stream>>>(deg, N_NODES, 0.0f);
  k_edge_enc<<<nblk((long)N_EDGES * EH), B, 0, stream>>>(eatt, ee_W, ee_b, ef2);
  k_deg<<<nblk(N_EDGES), B, 0, stream>>>(ei, deg);
  k_loop_acc<<<nblk((long)N_EDGES * EH), B, 0, stream>>>(ei, ef2);
  k_loop_div<<<nblk((long)N_NODES * EH), B, 0, stream>>>(deg, ef2);

  // --- node encoder: h0 = x @ ne_W^T + ne_b ---
  k_gemm_wmma<<<N_NODES / 16, B, 0, stream>>>(x, NFEAT, x, NFEAT, ne_W, NFEAT,
                                              ne_b, h0, 0);

  // --- two GATv2 layers ---
  for (int l = 0; l < 2; ++l) {
    const float* hin  = l ? h1 : h0;
    float*       hout = l ? h2 : h1;
    const float *Wl = gW[l][0], *bl = gW[l][1], *Wr = gW[l][2], *br = gW[l][3];
    const float *We = gW[l][4], *at = gW[l][5], *bi = gW[l][6];

    k_gemm_wmma<<<N_NODES / 16, B, 0, stream>>>(hin, HID, hin, HID, Wl, HID, bl, xl, 0);
    k_gemm_wmma<<<N_NODES / 16, B, 0, stream>>>(hin, HID, hin, HID, Wr, HID, br, xr, 0);

    k_fill<<<nblk((long)N_NODES * NHEAD), B, 0, stream>>>(emaxf, (long)N_NODES * NHEAD, 0.0f);
    k_fill<<<nblk((long)N_NODES * NHEAD), B, 0, stream>>>(esum,  (long)N_NODES * NHEAD, 0.0f);
    k_fill<<<nblk((long)N_NODES * HID), B, 0, stream>>>(hout, (long)N_NODES * HID, 0.0f);

    const long TE = (long)EP * NHEAD;
    k_scores<<<nblk(TE), B, 0, stream>>>(ei, ef2, xl, xr, We, at, alpha[l], emax);
    k_expsum<<<nblk(TE), B, 0, stream>>>(ei, alpha[l], emax, esum);
    k_aggregate<<<nblk(TE), B, 0, stream>>>(ei, alpha[l], esum, xl, hout);
    k_bias_relu<<<nblk((long)N_NODES * HID), B, 0, stream>>>(hout, bi);
  }

  // --- MLP head: mhid = relu([h0,h2] @ m1_W^T + m1_b); out = sigmoid(mhid @ m2_W^T + b) ---
  k_gemm_wmma<<<N_NODES / 16, B, 0, stream>>>(h0, HID, h2, HID, m1_W, 2 * HID,
                                              m1_b, mhid, 1);
  k_mlp_out<<<nblk(N_NODES), B, 0, stream>>>(mhid, m2_W, m2_b, out);

  (void)in_sizes; (void)n_in; (void)out_size; (void)ws_size;
}
